// MultiHeadSelfAttention_58617713656490
// MI455X (gfx1250) — matrix operand order repaired and checked by reading, not compiled, not run on hardware
//
#include <hip/hip_runtime.h>
#include <hip/hip_bf16.h>

typedef __bf16 bf16_t;
typedef __attribute__((ext_vector_type(16))) __bf16 v16bf;
typedef __attribute__((ext_vector_type(8)))  __bf16 v8bf;
typedef __attribute__((ext_vector_type(8)))  float  v8f;

#define B_    4
#define T_    2048
#define C_    1024
#define H_    16
#define D_    64
#define MTOT  (B_ * T_)          // 8192 rows

// ---------------------------------------------------------------------------
// WMMA fragment loaders (layouts per CDNA5 ISA 7.12.2, wave32)
// ---------------------------------------------------------------------------

// A fragment 16x32 bf16 from row-major [rows, ld].
__device__ inline v16bf load_a_bf16(const bf16_t* __restrict__ tile, int ld, int lane) {
  int row  = lane & 15;
  int koff = (lane >> 4) ? 8 : 0;
  const bf16_t* p = tile + (size_t)row * ld + koff;
  v8bf lo = *(const v8bf*)(p);        // K = koff + 0..7
  v8bf hi = *(const v8bf*)(p + 16);   // K = koff + 16..23
  return __builtin_shufflevector(lo, hi, 0,1,2,3,4,5,6,7,8,9,10,11,12,13,14,15);
}

// B fragment 32x16: B[k][n] = src[n0+n][k0+k], src rows contiguous.
__device__ inline v16bf load_b_rows_bf16(const bf16_t* __restrict__ src, int ld,
                                         int k0, int lane) {
  int n    = lane & 15;
  int koff = (lane >> 4) ? 8 : 0;
  const bf16_t* p = src + (size_t)n * ld + k0 + koff;
  v8bf lo = *(const v8bf*)(p);
  v8bf hi = *(const v8bf*)(p + 16);
  return __builtin_shufflevector(lo, hi, 0,1,2,3,4,5,6,7,8,9,10,11,12,13,14,15);
}

__device__ inline v8f wmma_bf16(v16bf a, v16bf b, v8f c) {
  return __builtin_amdgcn_wmma_f32_16x16x32_bf16(false, a, false, b, (short)0, c,
                                                 false, false);
}

// ---------------------------------------------------------------------------
// Double-buffered 32x64 GEMM core: acc[0..3] = rows m0..m0+15 x 4 col tiles,
// acc[4..7] = rows m0+16..m0+31. A: row-major [.,C], B: W rows (y = x W^T).
// ---------------------------------------------------------------------------
__device__ inline void gemm32x64(const bf16_t* __restrict__ Arow,
                                 const bf16_t* __restrict__ Brow,
                                 int lane, v8f acc[8]) {
  v16bf a0 = load_a_bf16(Arow, C_, lane);
  v16bf a1 = load_a_bf16(Arow + (size_t)16 * C_, C_, lane);
  v16bf b0 = load_b_rows_bf16(Brow, C_, 0, lane);
  v16bf b1 = load_b_rows_bf16(Brow + (size_t)16 * C_, C_, 0, lane);
  v16bf b2 = load_b_rows_bf16(Brow + (size_t)32 * C_, C_, 0, lane);
  v16bf b3 = load_b_rows_bf16(Brow + (size_t)48 * C_, C_, 0, lane);

  for (int k0 = 0; k0 < C_ - 32; k0 += 32) {
    int kn = k0 + 32;
    // prefetch next k-step into independent registers (loads overlap WMMAs)
    v16bf a0n = load_a_bf16(Arow + kn, C_, lane);
    v16bf a1n = load_a_bf16(Arow + (size_t)16 * C_ + kn, C_, lane);
    v16bf b0n = load_b_rows_bf16(Brow, C_, kn, lane);
    v16bf b1n = load_b_rows_bf16(Brow + (size_t)16 * C_, C_, kn, lane);
    v16bf b2n = load_b_rows_bf16(Brow + (size_t)32 * C_, C_, kn, lane);
    v16bf b3n = load_b_rows_bf16(Brow + (size_t)48 * C_, C_, kn, lane);

    acc[0] = wmma_bf16(a0, b0, acc[0]);
    acc[1] = wmma_bf16(a0, b1, acc[1]);
    acc[2] = wmma_bf16(a0, b2, acc[2]);
    acc[3] = wmma_bf16(a0, b3, acc[3]);
    acc[4] = wmma_bf16(a1, b0, acc[4]);
    acc[5] = wmma_bf16(a1, b1, acc[5]);
    acc[6] = wmma_bf16(a1, b2, acc[6]);
    acc[7] = wmma_bf16(a1, b3, acc[7]);

    a0 = a0n; a1 = a1n; b0 = b0n; b1 = b1n; b2 = b2n; b3 = b3n;
  }
  // tail k-step
  acc[0] = wmma_bf16(a0, b0, acc[0]);
  acc[1] = wmma_bf16(a0, b1, acc[1]);
  acc[2] = wmma_bf16(a0, b2, acc[2]);
  acc[3] = wmma_bf16(a0, b3, acc[3]);
  acc[4] = wmma_bf16(a1, b0, acc[4]);
  acc[5] = wmma_bf16(a1, b1, acc[5]);
  acc[6] = wmma_bf16(a1, b2, acc[6]);
  acc[7] = wmma_bf16(a1, b3, acc[7]);
}

// ---------------------------------------------------------------------------
// Kernel 0: bulk f32 -> bf16 conversion (8 elements / thread, vectorized)
// ---------------------------------------------------------------------------
__global__ __launch_bounds__(256)
void cvt_kernel(const float* __restrict__ src, bf16_t* __restrict__ dst, int n) {
  int i = (blockIdx.x * 256 + threadIdx.x) * 8;
  if (i >= n) return;
#pragma unroll
  for (int e = 0; e < 8; ++e) dst[i + e] = (__bf16)src[i + e];
}

// ---------------------------------------------------------------------------
// Kernel 1: fused QKV projection  y = x @ W^T + b
//   Q,K -> bf16 [B,H,T,D];  V -> bf16 transposed [B,H,D,T]
// block = 256 (8 waves: 2 row-groups x 4 col-groups), WG tile 64x256.
// grid = (1024/256, 8192/64, 3)
// ---------------------------------------------------------------------------
__global__ __launch_bounds__(256)
void qkv_kernel(const bf16_t* __restrict__ x,
                const bf16_t* __restrict__ wq, const float* __restrict__ bq,
                const bf16_t* __restrict__ wk, const float* __restrict__ bk,
                const bf16_t* __restrict__ wv, const float* __restrict__ bv,
                bf16_t* __restrict__ Q, bf16_t* __restrict__ K,
                bf16_t* __restrict__ Vt) {
  const bf16_t* w; const float* bias;
  if (blockIdx.z == 0)      { w = wq; bias = bq; }
  else if (blockIdx.z == 1) { w = wk; bias = bk; }
  else                      { w = wv; bias = bv; }

  int lane = threadIdx.x & 31;
  int wv8  = threadIdx.x >> 5;
  int wm   = wv8 & 1, wn = wv8 >> 1;
  int m0   = blockIdx.y * 64 + wm * 32;
  int n0   = blockIdx.x * 256 + wn * 64;

  v8f acc[8] = {};
  gemm32x64(x + (size_t)m0 * C_, w + (size_t)n0 * C_, lane, acc);

  int ncol = lane & 15;
#pragma unroll
  for (int rt = 0; rt < 2; ++rt) {
    int mbase = m0 + rt * 16 + ((lane >> 4) << 3);
#pragma unroll
    for (int c = 0; c < 4; ++c) {
      int n = n0 + c * 16 + ncol;
      int h = n >> 6, d = n & 63;
      float bb = bias[n];
#pragma unroll
      for (int r = 0; r < 8; ++r) {
        int m = mbase + r;
        int bidx = m >> 11;                // m / T
        int t    = m & (T_ - 1);
        bf16_t val = (__bf16)(acc[rt * 4 + c][r] + bb);
        if (blockIdx.z == 0)
          Q[(((size_t)(bidx * H_ + h)) * T_ + t) * D_ + d] = val;
        else if (blockIdx.z == 1)
          K[(((size_t)(bidx * H_ + h)) * T_ + t) * D_ + d] = val;
        else
          Vt[(((size_t)(bidx * H_ + h)) * D_ + d) * T_ + t] = val;  // transposed
      }
    }
  }
}

// ---------------------------------------------------------------------------
// Kernel 2: causal flash attention. One wave per 16-row query tile.
// grid = 1024 blocks of 8 waves.
// ---------------------------------------------------------------------------
__global__ __launch_bounds__(256)
void attn_kernel(const bf16_t* __restrict__ Q, const bf16_t* __restrict__ K,
                 const bf16_t* __restrict__ Vt, bf16_t* __restrict__ AO) {
  __shared__ bf16_t plds[8][16][40];     // 80B row stride: conflict-free, 16B-aligned

  int lane = threadIdx.x & 31;
  int wv8  = threadIdx.x >> 5;
  int gw   = blockIdx.x * 8 + wv8;
  int qt   = gw & 127;
  int bh   = gw >> 7;
  int qbase = qt * 16;

  const bf16_t* Qb  = Q  + (size_t)bh * T_ * D_;
  const bf16_t* Kb  = K  + (size_t)bh * T_ * D_;
  const bf16_t* Vtb = Vt + (size_t)bh * D_ * T_;

  v16bf qa0 = load_a_bf16(Qb + (size_t)qbase * D_, D_, lane);       // d 0..31
  v16bf qa1 = load_a_bf16(Qb + (size_t)qbase * D_ + 32, D_, lane);  // d 32..63

  v8f o[4] = {};
  float mrow[8], lrow[8];
#pragma unroll
  for (int r = 0; r < 8; ++r) { mrow[r] = -3.0e38f; lrow[r] = 0.f; }

  int qrow_lo = qbase + ((lane >> 4) << 3);
  int ncol    = lane & 15;

  for (int jb = 0; jb <= qbase + 15; jb += 32) {
    // ---- K fragments, then V fragments (V latency hides under softmax) ----
    v16bf kb00 = load_b_rows_bf16(Kb + (size_t)jb * D_, D_, 0, lane);
    v16bf kb01 = load_b_rows_bf16(Kb + (size_t)jb * D_, D_, 32, lane);
    v16bf kb10 = load_b_rows_bf16(Kb + (size_t)(jb + 16) * D_, D_, 0, lane);
    v16bf kb11 = load_b_rows_bf16(Kb + (size_t)(jb + 16) * D_, D_, 32, lane);
    v16bf vf0 = load_b_rows_bf16(Vtb,                     T_, jb, lane);
    v16bf vf1 = load_b_rows_bf16(Vtb + (size_t)16 * T_,   T_, jb, lane);
    v16bf vf2 = load_b_rows_bf16(Vtb + (size_t)32 * T_,   T_, jb, lane);
    v16bf vf3 = load_b_rows_bf16(Vtb + (size_t)48 * T_,   T_, jb, lane);

    v8f s0 = {}, s1 = {};
    s0 = wmma_bf16(qa0, kb00, s0);
    s0 = wmma_bf16(qa1, kb01, s0);
    s1 = wmma_bf16(qa0, kb10, s1);
    s1 = wmma_bf16(qa1, kb11, s1);

    // ---- online softmax (a row = one VGPR across 16 lanes) ----
    float p0[8], p1[8];
#pragma unroll
    for (int r = 0; r < 8; ++r) {
      int row = qrow_lo + r;
      float v0 = (jb + ncol      <= row) ? s0[r] * 0.125f : -3.0e38f;
      float v1 = (jb + 16 + ncol <= row) ? s1[r] * 0.125f : -3.0e38f;
      float mx = fmaxf(v0, v1);
#pragma unroll
      for (int sh = 1; sh < 16; sh <<= 1)
        mx = fmaxf(mx, __shfl_xor(mx, sh, 32));
      float mnew = fmaxf(mrow[r], mx);
      float corr = __expf(mrow[r] - mnew);
      mrow[r] = mnew;
      float e0 = __expf(v0 - mnew);
      float e1 = __expf(v1 - mnew);
      p0[r] = e0; p1[r] = e1;
      float rs = e0 + e1;
#pragma unroll
      for (int sh = 1; sh < 16; sh <<= 1)
        rs += __shfl_xor(rs, sh, 32);
      lrow[r] = lrow[r] * corr + rs;
      o[0][r] *= corr; o[1][r] *= corr; o[2][r] *= corr; o[3][r] *= corr;
    }

    // ---- P: C/D-layout -> LDS -> A-layout (transpose bounce) ----
#pragma unroll
    for (int r = 0; r < 8; ++r) {
      int row = ((lane >> 4) << 3) + r;
      plds[wv8][row][ncol]      = (__bf16)p0[r];
      plds[wv8][row][16 + ncol] = (__bf16)p1[r];
    }
    asm volatile("s_wait_dscnt 0" ::: "memory");  // wave-local LDS RAW fence
    v16bf pa = load_a_bf16(&plds[wv8][0][0], 40, lane);

    // ---- O += P @ V ----
    o[0] = wmma_bf16(pa, vf0, o[0]);
    o[1] = wmma_bf16(pa, vf1, o[1]);
    o[2] = wmma_bf16(pa, vf2, o[2]);
    o[3] = wmma_bf16(pa, vf3, o[3]);
  }

  // ---- normalize + write merged-head bf16 output [B,T,C] ----
  int b = bh >> 4, h = bh & 15;
  float inv[8];
#pragma unroll
  for (int r = 0; r < 8; ++r) inv[r] = 1.0f / lrow[r];
#pragma unroll
  for (int cg = 0; cg < 4; ++cg) {
#pragma unroll
    for (int r = 0; r < 8; ++r) {
      int t   = qbase + ((lane >> 4) << 3) + r;
      int col = h * 64 + cg * 16 + ncol;
      AO[((size_t)(b * T_ + t)) * C_ + col] = (__bf16)(o[cg][r] * inv[r]);
    }
  }
}

// ---------------------------------------------------------------------------
// Kernel 3: output projection  out = AO @ wo^T + bo  (f32)   grid = (4,128)
// ---------------------------------------------------------------------------
__global__ __launch_bounds__(256)
void proj_kernel(const bf16_t* __restrict__ A, const bf16_t* __restrict__ wo,
                 const float* __restrict__ bo, float* __restrict__ out) {
  int lane = threadIdx.x & 31;
  int wv8  = threadIdx.x >> 5;
  int wm   = wv8 & 1, wn = wv8 >> 1;
  int m0   = blockIdx.y * 64 + wm * 32;
  int n0   = blockIdx.x * 256 + wn * 64;

  v8f acc[8] = {};
  gemm32x64(A + (size_t)m0 * C_, wo + (size_t)n0 * C_, lane, acc);

  int ncol = lane & 15;
#pragma unroll
  for (int rt = 0; rt < 2; ++rt) {
    int mbase = m0 + rt * 16 + ((lane >> 4) << 3);
#pragma unroll
    for (int c = 0; c < 4; ++c) {
      int n = n0 + c * 16 + ncol;
      float bb = bo[n];
#pragma unroll
      for (int r = 0; r < 8; ++r) {
        int m = mbase + r;
        out[(size_t)m * C_ + n] = acc[rt * 4 + c][r] + bb;
      }
    }
  }
}

// ---------------------------------------------------------------------------
extern "C" void kernel_launch(void* const* d_in, const int* in_sizes, int n_in,
                              void* d_out, int out_size, void* d_ws, size_t ws_size,
                              hipStream_t stream) {
  const float* x  = (const float*)d_in[0];
  const float* wq = (const float*)d_in[1];
  const float* bq = (const float*)d_in[2];
  const float* wk = (const float*)d_in[3];
  const float* bk = (const float*)d_in[4];
  const float* wv = (const float*)d_in[5];
  const float* bv = (const float*)d_in[6];
  const float* wo = (const float*)d_in[7];
  const float* bo = (const float*)d_in[8];
  float* out = (float*)d_out;

  const size_t xe = (size_t)MTOT * C_;      // 8M
  const size_t we = (size_t)C_ * C_;        // 1M
  bf16_t* xbf = (bf16_t*)d_ws;
  bf16_t* wqb = xbf + xe;
  bf16_t* wkb = wqb + we;
  bf16_t* wvb = wkb + we;
  bf16_t* wob = wvb + we;
  bf16_t* Q   = wob + we;
  bf16_t* K   = Q + xe;
  bf16_t* Vt  = K + xe;
  bf16_t* AO  = Vt + xe;

  // 0) one-time f32 -> bf16 conversions
  cvt_kernel<<<dim3((int)(xe / 8 / 256)), 256, 0, stream>>>(x, xbf, (int)xe);
  cvt_kernel<<<dim3((int)(we / 8 / 256)), 256, 0, stream>>>(wq, wqb, (int)we);
  cvt_kernel<<<dim3((int)(we / 8 / 256)), 256, 0, stream>>>(wk, wkb, (int)we);
  cvt_kernel<<<dim3((int)(we / 8 / 256)), 256, 0, stream>>>(wv, wvb, (int)we);
  cvt_kernel<<<dim3((int)(we / 8 / 256)), 256, 0, stream>>>(wo, wob, (int)we);

  // 1) QKV projections (z = 0/1/2 -> q/k/v),  V stored transposed
  qkv_kernel<<<dim3(C_ / 256, MTOT / 64, 3), 256, 0, stream>>>(
      xbf, wqb, bq, wkb, bk, wvb, bv, Q, K, Vt);

  // 2) causal flash attention
  attn_kernel<<<dim3((B_ * H_ * (T_ / 16)) / 8, 1, 1), 256, 0, stream>>>(Q, K, Vt, AO);

  // 3) output projection
  proj_kernel<<<dim3(C_ / 256, MTOT / 64, 1), 256, 0, stream>>>(AO, wob, bo, out);
}
